// ADConv_532575945331
// MI455X (gfx1250) — matrix operand order repaired and checked by reading, not compiled, not run on hardware
//
#include <hip/hip_runtime.h>
#include <hip/hip_bf16.h>
#include <math.h>

// ---------------- problem constants ----------------
#define NB    8
#define HH    56
#define WW    56
#define HW    3136          // 56*56
#define NHW   25088         // 8*3136
#define CIN   64
#define INTER 64
#define BS    108           // conv2 out channels
#define BSP   112           // padded to 7*16
#define TT    18
#define MM    6
#define LL    49            // 7x7 window
#define COUT  128
#define KTOT  576           // 64*9
#define KC    384           // 64*6

// padded activation geometry (NHWC bf16, zero halo baked in)
#define XP    62            // 56 + 2*3  (serves 3x3 conv pad1 and 7x7 fusion pad3)
#define XPP   3844          // 62*62
#define YP    58            // 56 + 2*1
#define YPP   3364          // 58*58

typedef __attribute__((ext_vector_type(16))) __bf16 v16bf;
typedef __attribute__((ext_vector_type(8)))  float  v8f;

union FragU { uint4 u[2]; v16bf v; };

static __device__ inline unsigned pk2(float a, float b) {
    unsigned short lo = __builtin_bit_cast(unsigned short, (__bf16)a);
    unsigned short hi = __builtin_bit_cast(unsigned short, (__bf16)b);
    return (unsigned)lo | ((unsigned)hi << 16);
}

// ---------------- prep kernels ----------------
// x [N][C][H][W] f32 -> xbp [N][62][62][64] bf16 with zero pad-3 halo
__global__ void k_prep_xbp(const float* __restrict__ x, __bf16* __restrict__ xbp, int total) {
    int i = blockIdx.x * blockDim.x + threadIdx.x;
    if (i >= total) return;
    int pixp = i >> 6, c = i & 63;
    int n = pixp / XPP, rem = pixp % XPP;
    int h = rem / XP, w = rem % XP;
    int hi = h - 3, wi = w - 3;
    float v = 0.f;
    if ((unsigned)hi < (unsigned)HH && (unsigned)wi < (unsigned)WW)
        v = x[((size_t)(n * 64 + c)) * HW + hi * WW + wi];
    xbp[i] = (__bf16)v;
}

// conv w [OC][64][3][3] f32 -> wb [OCpad][576] bf16, k = tap*64 + ci
__global__ void k_prep_w(const float* __restrict__ wsrc, __bf16* __restrict__ wdst,
                         int OC, int OCpad) {
    int i = blockIdx.x * blockDim.x + threadIdx.x;
    if (i >= OCpad * KTOT) return;
    int o = i / KTOT, k = i % KTOT;
    int tap = k >> 6, ci = k & 63;
    wdst[i] = (o < OC) ? (__bf16)wsrc[((size_t)(o * 64 + ci)) * 9 + tap] : (__bf16)0.0f;
}

__global__ void k_prep_coef(const float* __restrict__ c, __bf16* __restrict__ cb, int total) {
    int i = blockIdx.x * blockDim.x + threadIdx.x;
    if (i < total) cb[i] = (__bf16)c[i];
}

__global__ void k_prep_bias2(const float* __restrict__ b, float* __restrict__ bp) {
    int i = threadIdx.x;
    if (i < BSP) bp[i] = (i < BS) ? b[i] : 0.0f;
}

// ---------------- implicit-GEMM 3x3 conv via WMMA bf16 ----------------
// xpad: NHWC bf16 with zero halo, row width Wp, center offset pad.
// wb: [OCpad][576] bf16, k = (ky*3+kx)*64 + ci. outraw: [OCpad][NHW] f32.
// blockDim = 32 * (OCpad/16). Each wave: 16 out-ch x 64 pixels (4 WMMA tiles).
__global__ void conv3x3_wmma(const __bf16* __restrict__ xpad,
                             const __bf16* __restrict__ wb,
                             const float* __restrict__ bias,
                             float* __restrict__ outraw,
                             int Wp, int pad) {
    const int lane = threadIdx.x & 31;
    const int wv   = threadIdx.x >> 5;
    const int ocb  = wv * 16;
    const int pixbase = blockIdx.x * 64;      // 3136 % 64 == 0: never crosses image
    const int n   = pixbase / HW;
    const int hw0 = pixbase % HW;
    const int p   = lane & 15;
    const int koff = (lane & 16) ? 16 : 0;    // B: K half per lane group
    const int asel = (lane & 16) ? 16 : 0;    // A: K half per lane group
    const __bf16* wrow = wb + (size_t)(ocb + p) * KTOT;

    int base[4];
    #pragma unroll
    for (int s = 0; s < 4; ++s) {
        int hw = hw0 + s * 16 + p;
        int h = hw / WW, w = hw % WW;
        base[s] = ((n * Wp + h + pad) * Wp + (w + pad)) * 64 + koff;
    }

    v8f acc[4] = {{}, {}, {}, {}};
    for (int kc = 0; kc < 18; ++kc) {
        const int kb  = kc * 32;
        const int tap = kb >> 6;
        const int cb  = kb & 63;
        const int dy = tap / 3 - 1, dx = tap % 3 - 1;
        const int toff = (dy * Wp + dx) * 64 + cb;

        FragU afr;
        afr.u[0] = *(const uint4*)(wrow + kb + asel);
        afr.u[1] = *(const uint4*)(wrow + kb + 8 + asel);

        FragU bfr[4];
        #pragma unroll
        for (int s = 0; s < 4; ++s) {
            const uint4* src = (const uint4*)(xpad + base[s] + toff);
            bfr[s].u[0] = src[0]; bfr[s].u[1] = src[1];
        }
        #pragma unroll
        for (int s = 0; s < 4; ++s)
            acc[s] = __builtin_amdgcn_wmma_f32_16x16x32_bf16(
                false, afr.v, false, bfr[s].v, (short)0, acc[s], false, false);
    }

    const int rowoff = (lane & 16) ? 8 : 0;
    #pragma unroll
    for (int s = 0; s < 4; ++s) {
        #pragma unroll
        for (int v = 0; v < 8; ++v) {
            int row = ocb + v + rowoff;
            outraw[(size_t)row * NHW + pixbase + s * 16 + p] = acc[s][v] + bias[row];
        }
    }
}

// ---------------- BN (training-mode batch stats) ----------------
__global__ void k_bnstats(const float* __restrict__ raw,
                          const float* __restrict__ g, const float* __restrict__ b,
                          float* __restrict__ ss) {
    const int ch = blockIdx.x;
    const float* pr = raw + (size_t)ch * NHW;
    float s = 0.f, s2 = 0.f;
    for (int i = threadIdx.x; i < NHW; i += blockDim.x) {
        float v = pr[i]; s += v; s2 += v * v;
    }
    __shared__ float sh[256], sh2[256];
    sh[threadIdx.x] = s; sh2[threadIdx.x] = s2;
    __syncthreads();
    for (int o = 128; o > 0; o >>= 1) {
        if (threadIdx.x < o) { sh[threadIdx.x] += sh[threadIdx.x + o]; sh2[threadIdx.x] += sh2[threadIdx.x + o]; }
        __syncthreads();
    }
    if (threadIdx.x == 0) {
        float mean = sh[0] * (1.0f / NHW);
        float var  = sh2[0] * (1.0f / NHW) - mean * mean;
        float sc = g[ch] * rsqrtf(var + 1e-5f);
        ss[2 * ch]     = sc;
        ss[2 * ch + 1] = b[ch] - mean * sc;
    }
}

// BN+tanh: [C][NHW] f32 -> padded NHWC bf16 [N][58][58][64] with zero halo
__global__ void k_bnapply_pad(const float* __restrict__ raw, const float* __restrict__ ss,
                              __bf16* __restrict__ ypad, int total) {
    int i = blockIdx.x * blockDim.x + threadIdx.x;
    if (i >= total) return;
    int pixp = i >> 6, ch = i & 63;
    int n = pixp / YPP, rem = pixp % YPP;
    int h = rem / YP, w = rem % YP;
    int hi = h - 1, wi = w - 1;
    float v = 0.f;
    if ((unsigned)hi < (unsigned)HH && (unsigned)wi < (unsigned)WW) {
        int pix = n * HW + hi * WW + wi;
        v = tanhf(raw[(size_t)ch * NHW + pix] * ss[2 * ch] + ss[2 * ch + 1]);
    }
    ypad[i] = (__bf16)v;
}

// BN+tanh -> b_pred [pix][108] f32
__global__ void k_bnapply_f32(const float* __restrict__ raw, const float* __restrict__ ss,
                              float* __restrict__ outf, int total) {
    int i = blockIdx.x * blockDim.x + threadIdx.x;
    if (i >= total) return;
    int ch = i / NHW, pix = i % NHW;
    float v = raw[i] * ss[2 * ch] + ss[2 * ch + 1];
    outf[(size_t)pix * BS + ch] = tanhf(v);
}

// ---------------- fused basis expansion + window contraction ----------------
// per pixel: b_full[6][49] = b_pred[6][18] x bases[18][49];
// bases_out[c*6+m] = sum_l x[c, tap l] * b_full[m][l]  -> boutb [pix][384] bf16
__global__ void k_fuse(const __bf16* __restrict__ xpad,
                       const float* __restrict__ bpred,
                       const float* __restrict__ bases,
                       __bf16* __restrict__ boutb) {
    __shared__ float basesS[TT * LL];      // 882
    __shared__ float bpredS[8][BS];        // 8 waves x 108
    __shared__ float bfullS[8][MM * LL];   // 8 x 294

    const int tid  = threadIdx.x;
    const int lane = tid & 31;
    const int wv   = tid >> 5;
    const int pix  = blockIdx.x * 8 + wv;

    for (int i = tid; i < TT * LL; i += 256) basesS[i] = bases[i];
    for (int j = lane; j < BS; j += 32) bpredS[wv][j] = bpred[(size_t)pix * BS + j];
    __syncthreads();

    for (int i = lane; i < MM * LL; i += 32) {
        int m = i / LL, l = i % LL;
        float s = 0.f;
        #pragma unroll
        for (int t = 0; t < TT; ++t) s += bpredS[wv][m * TT + t] * basesS[t * LL + l];
        bfullS[wv][i] = s;
    }
    __syncthreads();

    const int n = pix / HW, hw = pix % HW;
    const int h = hw / WW, w = hw % WW;
    float a0[MM], a1[MM];
    #pragma unroll
    for (int m = 0; m < MM; ++m) { a0[m] = 0.f; a1[m] = 0.f; }

    // padded base: unpadded (h,w) maps to padded (h+3,w+3); tap l offsets (l/7, l%7)
    const int pbase = (n * XP + h) * XP + w;
    for (int l = 0; l < LL; ++l) {
        int off = (pbase + (l / 7) * XP + (l % 7)) * 64;
        float x0 = (float)xpad[off + lane];
        float x1 = (float)xpad[off + lane + 32];
        #pragma unroll
        for (int m = 0; m < MM; ++m) {
            float bfv = bfullS[wv][m * LL + l];
            a0[m] = fmaf(x0, bfv, a0[m]);
            a1[m] = fmaf(x1, bfv, a1[m]);
        }
    }
    unsigned* ob = (unsigned*)boutb + (size_t)pix * (KC / 2);
    ob[lane * 3 + 0]      = pk2(a0[0], a0[1]);
    ob[lane * 3 + 1]      = pk2(a0[2], a0[3]);
    ob[lane * 3 + 2]      = pk2(a0[4], a0[5]);
    ob[lane * 3 + 96 + 0] = pk2(a1[0], a1[1]);
    ob[lane * 3 + 96 + 1] = pk2(a1[2], a1[3]);
    ob[lane * 3 + 96 + 2] = pk2(a1[4], a1[5]);
}

// ---------------- final 1x1 conv: out = coef[128][384] x bases_out^T ----------------
// 8 waves = 128 out-ch; each wave 16 out-ch x 64 pixels (4 WMMA tiles)
__global__ void gemm_coef_wmma(const __bf16* __restrict__ boutb,
                               const __bf16* __restrict__ coefb,
                               float* __restrict__ out) {
    const int lane = threadIdx.x & 31;
    const int wv   = threadIdx.x >> 5;
    const int ocb  = wv * 16;
    const int pixbase = blockIdx.x * 64;
    const int p = lane & 15;
    const int koff = (lane & 16) ? 16 : 0;
    const int asel = (lane & 16) ? 16 : 0;
    const __bf16* arow = coefb + (size_t)(ocb + p) * KC;
    const __bf16* bcol[4];
    #pragma unroll
    for (int s = 0; s < 4; ++s)
        bcol[s] = boutb + (size_t)(pixbase + s * 16 + p) * KC + koff;

    v8f acc[4] = {{}, {}, {}, {}};
    for (int kc = 0; kc < 12; ++kc) {
        const int kb = kc * 32;
        if (kc < 11) __builtin_prefetch(bcol[0] + kb + 32, 0, 1);
        FragU afr;
        afr.u[0] = *(const uint4*)(arow + kb + asel);
        afr.u[1] = *(const uint4*)(arow + kb + 8 + asel);
        FragU bfr[4];
        #pragma unroll
        for (int s = 0; s < 4; ++s) {
            const uint4* bs = (const uint4*)(bcol[s] + kb);
            bfr[s].u[0] = bs[0]; bfr[s].u[1] = bs[1];
        }
        #pragma unroll
        for (int s = 0; s < 4; ++s)
            acc[s] = __builtin_amdgcn_wmma_f32_16x16x32_bf16(
                false, afr.v, false, bfr[s].v, (short)0, acc[s], false, false);
    }
    const int n = pixbase / HW;
    const int rowoff = (lane & 16) ? 8 : 0;
    #pragma unroll
    for (int s = 0; s < 4; ++s) {
        int hw = pixbase % HW + s * 16 + p;
        #pragma unroll
        for (int v = 0; v < 8; ++v) {
            int row = ocb + v + rowoff;
            out[((size_t)(n * COUT + row)) * HW + hw] = acc[s][v];
        }
    }
}

// ---------------- host launcher ----------------
static inline size_t alignup(size_t x) { return (x + 255) & ~(size_t)255; }

extern "C" void kernel_launch(void* const* d_in, const int* in_sizes, int n_in,
                              void* d_out, int out_size, void* d_ws, size_t ws_size,
                              hipStream_t stream) {
    (void)in_sizes; (void)n_in; (void)out_size; (void)ws_size;
    const float* x       = (const float*)d_in[0];
    const float* conv1_w = (const float*)d_in[1];
    const float* conv1_b = (const float*)d_in[2];
    const float* bn1_g   = (const float*)d_in[3];
    const float* bn1_b   = (const float*)d_in[4];
    const float* conv2_w = (const float*)d_in[5];
    const float* conv2_b = (const float*)d_in[6];
    const float* bn2_g   = (const float*)d_in[7];
    const float* bn2_b   = (const float*)d_in[8];
    const float* coef    = (const float*)d_in[9];
    const float* bases   = (const float*)d_in[10];
    float* out = (float*)d_out;

    char* ws = (char*)d_ws;
    size_t cur = 0;
    auto take = [&](size_t bytes) { size_t off = cur; cur = alignup(cur + bytes); return off; };

    __bf16* xbp    = (__bf16*)(ws + take((size_t)NB * XPP * 64 * 2));  // pad-3 NHWC x
    __bf16* wb1    = (__bf16*)(ws + take((size_t)INTER * KTOT * 2));
    __bf16* wb2    = (__bf16*)(ws + take((size_t)BSP * KTOT * 2));
    __bf16* coefb  = (__bf16*)(ws + take((size_t)COUT * KC * 2));
    float*  bias2p = (float*) (ws + take((size_t)BSP * 4));
    float*  ss1    = (float*) (ws + take((size_t)INTER * 2 * 4));
    float*  ss2    = (float*) (ws + take((size_t)BS * 2 * 4));
    float*  y1raw  = (float*) (ws + take((size_t)INTER * NHW * 4));
    float*  y2raw  = (float*) (ws + take((size_t)BSP * NHW * 4));
    __bf16* y1bp   = (__bf16*)(ws + take((size_t)NB * YPP * 64 * 2));  // pad-1 NHWC y1
    float*  bpred  = (float*) (ws + take((size_t)NHW * BS * 4));
    __bf16* boutb  = (__bf16*)(ws + take((size_t)NHW * KC * 2));

    const int B = 256;
    const int xbp_total = NB * XPP * 64;
    const int y1p_total = NB * YPP * 64;

    // prep
    k_prep_xbp <<<(xbp_total + B - 1) / B, B, 0, stream>>>(x, xbp, xbp_total);
    k_prep_w   <<<(INTER * KTOT + B - 1) / B, B, 0, stream>>>(conv1_w, wb1, INTER, INTER);
    k_prep_w   <<<(BSP * KTOT + B - 1) / B, B, 0, stream>>>(conv2_w, wb2, BS, BSP);
    k_prep_coef<<<(COUT * KC + B - 1) / B, B, 0, stream>>>(coef, coefb, COUT * KC);
    k_prep_bias2<<<1, 128, 0, stream>>>(conv2_b, bias2p);

    // conv1 -> BN -> tanh (into pad-1 NHWC bf16)
    conv3x3_wmma<<<NHW / 64, 32 * (INTER / 16), 0, stream>>>(xbp, wb1, conv1_b, y1raw, XP, 3);
    k_bnstats   <<<INTER, 256, 0, stream>>>(y1raw, bn1_g, bn1_b, ss1);
    k_bnapply_pad<<<(y1p_total + B - 1) / B, B, 0, stream>>>(y1raw, ss1, y1bp, y1p_total);

    // conv2 -> BN -> tanh  (b_pred)
    conv3x3_wmma<<<NHW / 64, 32 * (BSP / 16), 0, stream>>>(y1bp, wb2, bias2p, y2raw, YP, 1);
    k_bnstats   <<<BS, 256, 0, stream>>>(y2raw, bn2_g, bn2_b, ss2);
    k_bnapply_f32<<<((size_t)BS * NHW + B - 1) / B, B, 0, stream>>>(y2raw, ss2, bpred, BS * NHW);

    // fused basis projection + window contraction
    k_fuse<<<NHW / 8, 256, 0, stream>>>(xbp, bpred, bases, boutb);

    // final 1x1 conv as WMMA GEMM
    gemm_coef_wmma<<<NHW / 64, 256, 0, stream>>>(boutb, coefb, out);
}
